// Predictor_35424890258179
// MI455X (gfx1250) — matrix operand order repaired and checked by reading, not compiled, not run on hardware
//
#include <hip/hip_runtime.h>
#include <hip/hip_bf16.h>

// CDNA5 / gfx1250. Dense GEMMs via v_wmma_f32_16x16x32_bf16 (fp32 accum),
// sparse aggregation via native f32 global atomics. Memory floor ~18us
// (TSdata 410MB @ 23.3TB/s); intermediates are L2-resident (192MB L2).

typedef __attribute__((ext_vector_type(16))) __bf16 v16bf;
typedef __attribute__((ext_vector_type(8)))  float  v8f;

#define TAU_INV (1.0f / 3.0f)
#define BETA_C  0.5f

__device__ __forceinline__ void atomAddF(float* p, float v) {
  __hip_atomic_fetch_add(p, v, __ATOMIC_RELAXED, __HIP_MEMORY_SCOPE_AGENT);
}

// ---------------- elementwise helpers ----------------
__global__ void k_fill(float* __restrict__ p, float v, int n) {
  int i = blockIdx.x * blockDim.x + threadIdx.x;
  if (i < n) p[i] = v;
}

// deg[dst[e]] += w[e]   (self-loop "1" pre-filled by k_fill)
__global__ void k_deg(const int* __restrict__ dst, const float* __restrict__ w,
                      float* __restrict__ deg, int e) {
  int i = blockIdx.x * blockDim.x + threadIdx.x;
  if (i < e) atomAddF(deg + dst[i], w[i]);
}

__global__ void k_dinv(const float* __restrict__ deg, float* __restrict__ dinv, int n) {
  int i = blockIdx.x * blockDim.x + threadIdx.x;
  if (i < n) { float d = deg[i]; dinv[i] = d > 0.f ? rsqrtf(d) : 0.f; }
}

// ---------------- weight pre-pack to WMMA B-fragment layout (bf16) ----------
// B element e at lane L holds W[kb*32 + (L/16)*8 + (e%8) + (e/8)*16][nt*16 + (L%16)]
__global__ void k_prepack(const float* __restrict__ W, __bf16* __restrict__ dst,
                          int nkb, int nnt, int ncols) {
  int idx = blockIdx.x * blockDim.x + threadIdx.x;
  int total = nkb * nnt * 32 * 16;
  if (idx >= total) return;
  int e    = idx & 15;
  int lane = (idx >> 4) & 31;
  int nt   = (idx >> 9) % nnt;
  int kb   = idx / (512 * nnt);
  int k = kb * 32 + ((lane >> 4) << 3) + (e & 7) + ((e >> 3) << 4);
  int n = nt * 16 + (lane & 15);
  dst[idx] = (__bf16)W[(size_t)k * ncols + n];
}

// Build A fragment (16-bit A layout, K-block of 32): caller bakes half*8 into ap.
__device__ __forceinline__ v16bf load_a32(const float* __restrict__ ap) {
  float4 a0 = *(const float4*)(ap);
  float4 a1 = *(const float4*)(ap + 4);
  float4 a2 = *(const float4*)(ap + 16);
  float4 a3 = *(const float4*)(ap + 20);
  v16bf A;
  A[0]=(__bf16)a0.x;  A[1]=(__bf16)a0.y;  A[2]=(__bf16)a0.z;  A[3]=(__bf16)a0.w;
  A[4]=(__bf16)a1.x;  A[5]=(__bf16)a1.y;  A[6]=(__bf16)a1.z;  A[7]=(__bf16)a1.w;
  A[8]=(__bf16)a2.x;  A[9]=(__bf16)a2.y;  A[10]=(__bf16)a2.z; A[11]=(__bf16)a2.w;
  A[12]=(__bf16)a3.x; A[13]=(__bf16)a3.y; A[14]=(__bf16)a3.z; A[15]=(__bf16)a3.w;
  return A;
}

// ---------------- GEMM1: x[N,128] -> xw_ts[N,64]+b_ts1, xw_g[N,64] ----------
__global__ void __launch_bounds__(256)
k_gemm1(const float* __restrict__ x, const v16bf* __restrict__ wf,
        const float* __restrict__ b_ts1,
        float* __restrict__ xw_ts, float* __restrict__ xw_g, int n) {
  const int lane = threadIdx.x & 31;
  const int wave = threadIdx.x >> 5;
  const int r0 = (blockIdx.x * 8 + wave) * 16;
  if (r0 >= n) return;                       // wave-uniform, EXEC stays full
  const int half = lane >> 4;
  int mc = r0 + (lane & 15); if (mc >= n) mc = n - 1;

  v8f acc_t[4], acc_g[4];
  #pragma unroll
  for (int nt = 0; nt < 4; ++nt) {
    float b = b_ts1[nt * 16 + (lane & 15)];
    #pragma unroll
    for (int r = 0; r < 8; ++r) { acc_t[nt][r] = b; acc_g[nt][r] = 0.f; }
  }

  #pragma unroll
  for (int kb = 0; kb < 4; ++kb) {
    v16bf A = load_a32(x + (size_t)mc * 128 + kb * 32 + half * 8);
    #pragma unroll
    for (int nt = 0; nt < 4; ++nt) {
      v16bf Bt = wf[(kb * 4 + nt) * 32 + lane];
      acc_t[nt] = __builtin_amdgcn_wmma_f32_16x16x32_bf16(
          false, A, false, Bt, (short)0, acc_t[nt], false, false);
      v16bf Bg = wf[512 + (kb * 4 + nt) * 32 + lane];
      acc_g[nt] = __builtin_amdgcn_wmma_f32_16x16x32_bf16(
          false, A, false, Bg, (short)0, acc_g[nt], false, false);
    }
  }
  // C/D layout: element r at lane L -> row (L/16)*8+r, col L%16
  #pragma unroll
  for (int r = 0; r < 8; ++r) {
    int row = r0 + half * 8 + r;
    if (row < n) {
      #pragma unroll
      for (int nt = 0; nt < 4; ++nt) {
        xw_ts[(size_t)row * 64 + nt * 16 + (lane & 15)] = acc_t[nt][r];
        xw_g [(size_t)row * 64 + nt * 16 + (lane & 15)] = acc_g[nt][r];
      }
    }
  }
}

// ---------------- GEMM2: ht1[N,64]@Wts2+b_ts2 -> htw[N,16]; hg1@Wg2 -> hgw --
__global__ void __launch_bounds__(256)
k_gemm2(const float* __restrict__ in_t, const float* __restrict__ in_g,
        const v16bf* __restrict__ wf, const float* __restrict__ b_ts2,
        float* __restrict__ out_t, float* __restrict__ out_g, int n) {
  const int lane = threadIdx.x & 31;
  const int wave = threadIdx.x >> 5;
  const int r0 = (blockIdx.x * 8 + wave) * 16;
  if (r0 >= n) return;
  const int half = lane >> 4;
  int mc = r0 + (lane & 15); if (mc >= n) mc = n - 1;

  v8f at, ag;
  float b = b_ts2[lane & 15];
  #pragma unroll
  for (int r = 0; r < 8; ++r) { at[r] = b; ag[r] = 0.f; }

  #pragma unroll
  for (int kb = 0; kb < 2; ++kb) {
    v16bf At = load_a32(in_t + (size_t)mc * 64 + kb * 32 + half * 8);
    v16bf Ag = load_a32(in_g + (size_t)mc * 64 + kb * 32 + half * 8);
    v16bf Bt = wf[kb * 32 + lane];
    at = __builtin_amdgcn_wmma_f32_16x16x32_bf16(false, At, false, Bt, (short)0, at, false, false);
    v16bf Bg = wf[64 + kb * 32 + lane];
    ag = __builtin_amdgcn_wmma_f32_16x16x32_bf16(false, Ag, false, Bg, (short)0, ag, false, false);
  }
  #pragma unroll
  for (int r = 0; r < 8; ++r) {
    int row = r0 + half * 8 + r;
    if (row < n) {
      out_t[(size_t)row * 16 + (lane & 15)] = at[r];
      out_g[(size_t)row * 16 + (lane & 15)] = ag[r];
    }
  }
}

// ---------------- layer-1 init: ht1 = 0; hg1 = b_g1 + dinv^2 * xw_g ---------
__global__ void k_init1(const float* __restrict__ xw_g, const float* __restrict__ dinv,
                        const float* __restrict__ b_g1,
                        float* __restrict__ ht1, float* __restrict__ hg1, int n) {
  int tid = blockIdx.x * blockDim.x + threadIdx.x;
  if (tid >= n * 16) return;
  int node = tid >> 4;
  int q = (tid & 15) << 2;
  float di = dinv[node], s = di * di;
  float4 xg = *(const float4*)(xw_g + (size_t)node * 64 + q);
  float4 bb = *(const float4*)(b_g1 + q);
  float4 h;
  h.x = bb.x + s * xg.x; h.y = bb.y + s * xg.y;
  h.z = bb.z + s * xg.z; h.w = bb.w + s * xg.w;
  *(float4*)(hg1 + (size_t)node * 64 + q) = h;
  *(float4*)(ht1 + (size_t)node * 64 + q) = make_float4(0.f, 0.f, 0.f, 0.f);
}

// layer-2 self/bias: acc += BETA*(b_g2 + dinv^2 * hgw)
__global__ void k_self2(const float* __restrict__ hgw, const float* __restrict__ dinv,
                        const float* __restrict__ b_g2, float* __restrict__ acc, int n) {
  int tid = blockIdx.x * blockDim.x + threadIdx.x;
  if (tid >= n * 4) return;
  int node = tid >> 2;
  int q = (tid & 3) << 2;
  float di = dinv[node], s = di * di;
  float4 h = *(const float4*)(hgw + (size_t)node * 16 + q);
  float4 bb = *(const float4*)(b_g2 + q);
  float* o = acc + (size_t)node * 16 + q;
  o[0] += BETA_C * (bb.x + s * h.x);
  o[1] += BETA_C * (bb.y + s * h.y);
  o[2] += BETA_C * (bb.z + s * h.z);
  o[3] += BETA_C * (bb.w + s * h.w);
}

// ---------------- generic edge scatter: out[d] += coef * sf[s] --------------
// coef = w[e]*scale  (* dinv[s]*dinv[d] if dinv != null).  4 feats per thread.
__global__ void k_scatter(const int* __restrict__ dsti, const int* __restrict__ srci,
                          const float* __restrict__ w, const float* __restrict__ dinv,
                          const float* __restrict__ sf, float* __restrict__ of,
                          float scale, int nnz, int F) {
  int per = F >> 2;
  long long tid = (long long)blockIdx.x * blockDim.x + threadIdx.x;
  long long e = tid / per;
  if (e >= nnz) return;
  int q = (int)(tid % per) << 2;
  int s = srci[e], d = dsti[e];
  float c = w[e] * scale;
  if (dinv) c *= dinv[s] * dinv[d];
  float4 v = *(const float4*)(sf + (size_t)s * F + q);
  float* o = of + (size_t)d * F + q;
  atomAddF(o + 0, c * v.x);
  atomAddF(o + 1, c * v.y);
  atomAddF(o + 2, c * v.z);
  atomAddF(o + 3, c * v.w);
}

// ---------------- softmax over C=16 classes, in place -----------------------
__global__ void k_softmax(float* __restrict__ out, int n) {
  int i = blockIdx.x * blockDim.x + threadIdx.x;
  if (i >= n) return;
  float* row = out + (size_t)i * 16;
  float v[16], m = -3.4e38f;
  #pragma unroll
  for (int c = 0; c < 16; ++c) { v[c] = row[c]; m = fmaxf(m, v[c]); }
  float sum = 0.f;
  #pragma unroll
  for (int c = 0; c < 16; ++c) { v[c] = __expf(v[c] - m); sum += v[c]; }
  float inv = 1.f / sum;
  #pragma unroll
  for (int c = 0; c < 16; ++c) row[c] = v[c] * inv;
}

extern "C" void kernel_launch(void* const* d_in, const int* in_sizes, int n_in,
                              void* d_out, int out_size, void* d_ws, size_t ws_size,
                              hipStream_t stream) {
  const float* TS   = (const float*)d_in[0];
  const int*   tedg = (const int*)  d_in[1];
  const float* tval = (const float*)d_in[2];
  const int*   edg  = (const int*)  d_in[3];
  const float* eatt = (const float*)d_in[4];
  const float* Wts1 = (const float*)d_in[5];
  const float* bts1 = (const float*)d_in[6];
  const float* Wts2 = (const float*)d_in[7];
  const float* bts2 = (const float*)d_in[8];
  const float* Wg1  = (const float*)d_in[9];
  const float* bg1  = (const float*)d_in[10];
  const float* Wg2  = (const float*)d_in[11];
  const float* bg2  = (const float*)d_in[12];
  float* out = (float*)d_out;

  const int C  = in_sizes[8];                 // 16
  const int H  = in_sizes[6];                 // 64
  const int Fi = in_sizes[5] / H;             // 128
  const int N  = out_size / C;                // 100000
  const int ET = in_sizes[1] / 2;             // 400000
  const int E  = in_sizes[3] / 2;             // 800000
  const int T  = (int)((long long)in_sizes[0] / ((long long)N * Fi)); // 8

  const int* trow = tedg;      const int* tcol = tedg + ET;
  const int* esrc = edg;       const int* edst = edg + E;

  // workspace carve (256B aligned)
  char* ws = (char*)d_ws;
  auto alloc = [&](size_t bytes) -> char* {
    char* p = ws; ws += (bytes + 255) & ~(size_t)255; return p;
  };
  float*  xw_ts = (float*)alloc((size_t)N * H * 4);
  float*  xw_g  = (float*)alloc((size_t)N * H * 4);
  float*  ht1   = (float*)alloc((size_t)N * H * 4);
  float*  hg1   = (float*)alloc((size_t)N * H * 4);
  float*  htw   = (float*)alloc((size_t)N * C * 4);
  float*  hgw   = (float*)alloc((size_t)N * C * 4);
  float*  deg   = (float*)alloc((size_t)N * 4);
  float*  dinv  = (float*)alloc((size_t)N * 4);
  __bf16* wf1   = (__bf16*)alloc((size_t)2 * 8192 * 2);
  __bf16* wf2   = (__bf16*)alloc((size_t)2 * 1024 * 2);

  const int BS = 256;
  auto cdiv = [](long long a, long long b) { return (int)((a + b - 1) / b); };

  // acc (= d_out) starts at zero
  k_fill<<<cdiv((long long)N * C, BS), BS, 0, stream>>>(out, 0.f, N * C);

  // pre-pack weights into bf16 B-fragment layout
  k_prepack<<<cdiv(8192, BS), BS, 0, stream>>>(Wts1, wf1,        4, 4, H);
  k_prepack<<<cdiv(8192, BS), BS, 0, stream>>>(Wg1,  wf1 + 8192, 4, 4, H);
  k_prepack<<<cdiv(1024, BS), BS, 0, stream>>>(Wts2, wf2,        2, 1, C);
  k_prepack<<<cdiv(1024, BS), BS, 0, stream>>>(Wg2,  wf2 + 1024, 2, 1, C);

  const int gemmBlocks = cdiv((long long)N, 16 * 8);

  for (int t = 0; t < T; ++t) {
    const float* x = TS   + (size_t)t * N * Fi;
    const float* w = eatt + (size_t)t * E;

    // degree (self-loop weight 1) -> dinv = rsqrt(deg)
    k_fill<<<cdiv(N, BS), BS, 0, stream>>>(deg, 1.0f, N);
    k_deg <<<cdiv(E, BS), BS, 0, stream>>>(edst, w, deg, E);
    k_dinv<<<cdiv(N, BS), BS, 0, stream>>>(deg, dinv, N);

    // layer 1: fused dual GEMM (WMMA bf16, fp32 accum)
    k_gemm1<<<gemmBlocks, 256, 0, stream>>>(x, (const v16bf*)wf1, bts1, xw_ts, xw_g, N);

    // ht1 = 0; hg1 = b_g1 + dinv^2 * xw_g  (GCN self-loop term)
    k_init1<<<cdiv((long long)N * 16, BS), BS, 0, stream>>>(xw_g, dinv, bg1, ht1, hg1, N);

    // ht1 += (val/tau) * xw_ts[tcol] ; hg1 += norm * xw_g[esrc]
    k_scatter<<<cdiv((long long)ET * (H / 4), BS), BS, 0, stream>>>(
        trow, tcol, tval, nullptr, xw_ts, ht1, TAU_INV, ET, H);
    k_scatter<<<cdiv((long long)E * (H / 4), BS), BS, 0, stream>>>(
        edst, esrc, w, dinv, xw_g, hg1, 1.0f, E, H);

    // layer 2: fused dual GEMM (64 -> 16)
    k_gemm2<<<gemmBlocks, 256, 0, stream>>>(ht1, hg1, (const v16bf*)wf2, bts2, htw, hgw, N);

    // acc += BETA*(b_g2 + dinv^2*hgw)  then sparse layer-2 contributions
    k_self2<<<cdiv((long long)N * 4, BS), BS, 0, stream>>>(hgw, dinv, bg2, out, N);
    k_scatter<<<cdiv((long long)ET * (C / 4), BS), BS, 0, stream>>>(
        trow, tcol, tval, nullptr, htw, out, TAU_INV, ET, C);
    k_scatter<<<cdiv((long long)E * (C / 4), BS), BS, 0, stream>>>(
        edst, esrc, w, dinv, hgw, out, BETA_C, E, C);
  }

  k_softmax<<<cdiv(N, BS), BS, 0, stream>>>(out, N);
}
